// MoERouter_24189255811637
// MI455X (gfx1250) — matrix operand order repaired and checked by reading, not compiled, not run on hardware
//
#include <hip/hip_runtime.h>
#include <hip/hip_bf16.h>

// ---------------------------------------------------------------------------
// MoE router for MI455X (gfx1250, wave32).
//
// Roofline: gate GEMM reads 268MB activations @23.3TB/s => ~11.5us floor,
// needs ~745 TFLOP/s to stay BW-bound. fp32->bf16 convert-in-register +
// v_wmma_f32_16x16x32_bf16 clears that; the cvt VALU work co-executes with
// XDL WMMA (16-bit WMMA tracked as TRANS per CDNA5 ISA 8.4).
//
// Data movement: X is streamed once -> non-temporal loads (keep 192MB L2 for
// W, which every block re-reads each K-step). A-stream prefetched 2 chunks
// ahead via global_prefetch_b8.
// ---------------------------------------------------------------------------

typedef __attribute__((ext_vector_type(16))) __bf16 v16bf;
typedef __attribute__((ext_vector_type(8)))  float  v8f;
typedef __attribute__((ext_vector_type(4)))  float  vf4;   // clang vector: NT-loadable

#define HDIM     4096
#define NEXP     64
#define ROWS_PER_BLOCK 32
#define SORT_CHUNK 512

// Build one bf16 A/B fragment (16 values) from two vf4 pairs.
__device__ __forceinline__ v16bf pack16(vf4 x0, vf4 x1, vf4 x2, vf4 x3) {
    v16bf r;
    r[0]=(__bf16)x0[0]; r[1]=(__bf16)x0[1]; r[2]=(__bf16)x0[2]; r[3]=(__bf16)x0[3];
    r[4]=(__bf16)x1[0]; r[5]=(__bf16)x1[1]; r[6]=(__bf16)x1[2]; r[7]=(__bf16)x1[3];
    r[8]=(__bf16)x2[0]; r[9]=(__bf16)x2[1]; r[10]=(__bf16)x2[2]; r[11]=(__bf16)x2[3];
    r[12]=(__bf16)x3[0]; r[13]=(__bf16)x3[1]; r[14]=(__bf16)x3[2]; r[15]=(__bf16)x3[3];
    return r;
}

// A fragment for K-chunk [kk, kk+32): ISA 7.12.2 "16-bit A-Matrix 16x32".
// lanes 0-15 : VGPR0-3 K=0..7,  VGPR4-7 K=16..23
// lanes 16-31: VGPR0-3 K=8..15, VGPR4-7 K=24..31
__device__ __forceinline__ v16bf loadA(const float* __restrict__ row, int kk, int hi) {
    const int ab = kk + hi * 8;
    // Non-temporal: each X row is consumed by exactly one block, once.
    vf4 a0 = __builtin_nontemporal_load(reinterpret_cast<const vf4*>(row + ab));
    vf4 a1 = __builtin_nontemporal_load(reinterpret_cast<const vf4*>(row + ab + 4));
    vf4 a2 = __builtin_nontemporal_load(reinterpret_cast<const vf4*>(row + ab + 16));
    vf4 a3 = __builtin_nontemporal_load(reinterpret_cast<const vf4*>(row + ab + 20));
    return pack16(a0, a1, a2, a3);
}

// B fragment (32x16 bf16, same K order as A):
// lanes 0-15 hold K=kk+0..7,kk+16..23 for N=l15; lanes 16-31 hold K=kk+8..15,kk+24..31.
__device__ __forceinline__ v16bf loadB(const float* __restrict__ row, int kk, int hi) {
    const int bb = kk + hi * 8;
    vf4 b0 = *reinterpret_cast<const vf4*>(row + bb);
    vf4 b1 = *reinterpret_cast<const vf4*>(row + bb + 4);
    vf4 b2 = *reinterpret_cast<const vf4*>(row + bb + 16);
    vf4 b3 = *reinterpret_cast<const vf4*>(row + bb + 20);
    return pack16(b0, b1, b2, b3);
}

// ---------------------------------------------------------------------------
// Kernel 1: gate GEMM (bf16 WMMA) + top-2 + 2-way softmax.
// Block: 256 threads = 8 waves tiling 2(M) x 4(N) 16x16 tiles => 32 tokens x
// 64 experts. K unrolled x2 with dual accumulators: 2 independent WMMAs per
// iteration (no D->C serial chain between them).
// ---------------------------------------------------------------------------
__global__ __launch_bounds__(256)
void moe_gate_wmma_kernel(const float* __restrict__ X,   // (T, 4096)
                          const float* __restrict__ W,   // (64, 4096)
                          float* __restrict__ outWts,    // (T*2)
                          float* __restrict__ outSelF,   // (T*2) expert ids as float
                          int*   __restrict__ flatExp)   // (T*2) expert ids (ws)
{
    const int tid   = threadIdx.x;
    const int wave  = tid >> 5;
    const int lane  = tid & 31;
    const int hi    = lane >> 4;
    const int l15   = lane & 15;
    const int mTile = wave >> 2;      // 0..1
    const int nTile = wave & 3;       // 0..3

    const int aRow = blockIdx.x * ROWS_PER_BLOCK + mTile * 16 + l15;
    const int bRow = nTile * 16 + l15;

    const float* __restrict__ aPtr = X + (size_t)aRow * HDIM;
    const float* __restrict__ bPtr = W + (size_t)bRow * HDIM;

    v8f acc0 = {};
    v8f acc1 = {};

    for (int kk = 0; kk < HDIM; kk += 64) {
        // Prefetch A two 64-wide chunks ahead (speculative; tail overrun OK).
        __builtin_prefetch(aPtr + kk + 128, 0, 0);

        v16bf a0 = loadA(aPtr, kk,      hi);
        v16bf b0 = loadB(bPtr, kk,      hi);
        v16bf a1 = loadA(aPtr, kk + 32, hi);
        v16bf b1 = loadB(bPtr, kk + 32, hi);

        acc0 = __builtin_amdgcn_wmma_f32_16x16x32_bf16(
                   false, a0, false, b0, (short)0, acc0, false, false);
        acc1 = __builtin_amdgcn_wmma_f32_16x16x32_bf16(
                   false, a1, false, b1, (short)0, acc1, false, false);
    }
    const v8f acc = acc0 + acc1;

    // C/D layout: lane L, VGPR r -> M = r + (L>=16 ? 8 : 0), N = L&15
    __shared__ float s_logits[ROWS_PER_BLOCK * NEXP];
    #pragma unroll
    for (int r = 0; r < 8; ++r)
        s_logits[(mTile * 16 + r + hi * 8) * NEXP + nTile * 16 + l15] = acc[r];
    __syncthreads();

    // Branchless per-token top-2 (v_cndmask chains, no exec-mask churn).
    // Renormalized top-2 softmax == 2-way softmax over the two top logits.
    if (tid < ROWS_PER_BLOCK) {
        const int token = blockIdx.x * ROWS_PER_BLOCK + tid;
        const float* lg = &s_logits[tid * NEXP];
        float best = lg[0], sec = -3.4e38f;
        int bi = 0, si = 0;
        #pragma unroll 8
        for (int e = 1; e < NEXP; ++e) {
            const float v = lg[e];
            const bool gtB = v > best;
            const bool gtS = v > sec;
            const float nsec = gtB ? best : (gtS ? v : sec);
            const int   nsi  = gtB ? bi   : (gtS ? e : si);
            best = gtB ? v : best;
            bi   = gtB ? e : bi;
            sec  = nsec;
            si   = nsi;
        }
        const float r  = __expf(sec - best);       // <= 1, no overflow
        const float p1 = 1.0f / (1.0f + r);
        outWts[token * 2]      = p1;
        outWts[token * 2 + 1]  = 1.0f - p1;
        outSelF[token * 2]     = (float)bi;
        outSelF[token * 2 + 1] = (float)si;
        flatExp[token * 2]     = bi;
        flatExp[token * 2 + 1] = si;
    }
}

// ---------------------------------------------------------------------------
// Stable counting sort of flat_experts (argsort permutation + bincount).
// Pass A: per-chunk expert histograms.
// ---------------------------------------------------------------------------
__global__ __launch_bounds__(256)
void moe_hist_kernel(const int* __restrict__ flatExp, int* __restrict__ hist, int TK)
{
    __shared__ int h[NEXP];
    if (threadIdx.x < NEXP) h[threadIdx.x] = 0;
    __syncthreads();
    const int start = blockIdx.x * SORT_CHUNK;
    for (int i = threadIdx.x; i < SORT_CHUNK; i += 256) {
        const int idx = start + i;
        if (idx < TK) atomicAdd(&h[flatExp[idx]], 1);
    }
    __syncthreads();
    if (threadIdx.x < NEXP) hist[blockIdx.x * NEXP + threadIdx.x] = h[threadIdx.x];
}

// Pass B: single block: m_sizes, expert exclusive prefix, per-(chunk,expert)
// base offsets (expert-major, chunk-minor => stable order).
__global__ __launch_bounds__(64)
void moe_scan_kernel(const int* __restrict__ hist, int* __restrict__ blockOff,
                     float* __restrict__ outMsz, int nblk)
{
    __shared__ int total[NEXP];
    __shared__ int base[NEXP];
    const int e = threadIdx.x;
    int s = 0;
    for (int b = 0; b < nblk; ++b) s += hist[b * NEXP + e];
    total[e]  = s;
    outMsz[e] = (float)s;
    __syncthreads();
    if (e == 0) {
        int run = 0;
        for (int i = 0; i < NEXP; ++i) { base[i] = run; run += total[i]; }
    }
    __syncthreads();
    int run = base[e];
    for (int b = 0; b < nblk; ++b) {
        blockOff[b * NEXP + e] = run;
        run += hist[b * NEXP + e];
    }
}

// Pass C: stable in-chunk rank, then scatter flat index to sorted position.
__global__ __launch_bounds__(256)
void moe_place_kernel(const int* __restrict__ flatExp, const int* __restrict__ blockOff,
                      float* __restrict__ outGather, int TK)
{
    __shared__ int ex[SORT_CHUNK];
    const int start = blockIdx.x * SORT_CHUNK;
    for (int i = threadIdx.x; i < SORT_CHUNK; i += 256) {
        const int idx = start + i;
        ex[i] = (idx < TK) ? flatExp[idx] : -1;
    }
    __syncthreads();
    for (int i = threadIdx.x; i < SORT_CHUNK; i += 256) {
        const int idx = start + i;
        if (idx < TK) {
            const int e = ex[i];
            int rank = 0;
            for (int j = 0; j < i; ++j) rank += (ex[j] == e) ? 1 : 0;
            const int pos = blockOff[blockIdx.x * NEXP + e] + rank;
            outGather[pos] = (float)idx;
        }
    }
}

// ---------------------------------------------------------------------------
extern "C" void kernel_launch(void* const* d_in, const int* in_sizes, int n_in,
                              void* d_out, int out_size, void* d_ws, size_t ws_size,
                              hipStream_t stream)
{
    const float* X = (const float*)d_in[0];   // (B*S, 4096) fp32
    const float* W = (const float*)d_in[1];   // (64, 4096)  fp32

    const int T  = in_sizes[0] / HDIM;        // 16384 tokens
    const int TK = T * 2;                     // top-2 flat items

    float* outF      = (float*)d_out;
    float* outWts    = outF;                  // [0,        TK)
    float* outSelF   = outF + TK;             // [TK,     2*TK)
    float* outMsz    = outF + 2 * TK;         // [2TK, 2TK+64)
    float* outGather = outF + 2 * TK + NEXP;  // [2TK+64, 3TK+64)

    const int nblk   = (TK + SORT_CHUNK - 1) / SORT_CHUNK;
    int* flatExp  = (int*)d_ws;               // TK ints
    int* hist     = flatExp + TK;             // nblk*64 ints
    int* blockOff = hist + nblk * NEXP;       // nblk*64 ints

    // 1) Gate GEMM + softmax/top-2/renorm (bandwidth-bound: 268MB stream).
    moe_gate_wmma_kernel<<<T / ROWS_PER_BLOCK, 256, 0, stream>>>(
        X, W, outWts, outSelF, flatExp);

    // 2) Stable counting-sort dispatch (tiny: 32768 items, 64 bins).
    moe_hist_kernel <<<nblk, 256, 0, stream>>>(flatExp, hist, TK);
    moe_scan_kernel <<<1,   64,  0, stream>>>(hist, blockOff, outMsz, nblk);
    moe_place_kernel<<<nblk, 256, 0, stream>>>(flatExp, blockOff, outGather, TK);
}
